// MultiHeadSelfAttention_75342316307218
// MI455X (gfx1250) — matrix operand order repaired and checked by reading, not compiled, not run on hardware
//
#include <hip/hip_runtime.h>
#include <math.h>

typedef __bf16 bf16_t;
typedef __attribute__((ext_vector_type(16))) __bf16 v16bf;
typedef __attribute__((ext_vector_type(8)))  __bf16 v8bf;
typedef __attribute__((ext_vector_type(8)))  float  v8f;

#define D_MODEL 512
#define SEQ     2048
#define BATCH   2
#define HEADS   8
#define DEPTH   64
#define NTOK    (BATCH * SEQ)   // 4096

union ABf { v16bf v; v8bf h[2]; };

static __device__ inline v8bf ld8(const bf16_t* p) {
  return *reinterpret_cast<const v8bf*>(p);
}

// Issue one 16-byte async copy global -> LDS (gfx1250 ASYNC path, ASYNCcnt).
// LDS address = low 32 bits of the generic address (ISA: LDS aperture keeps
// the LDS offset in addr[31:0]).
static __device__ inline void async_cp16(const bf16_t* gsrc, bf16_t* ldst) {
  unsigned int laddr = (unsigned int)(unsigned long long)ldst;
  asm volatile("global_load_async_to_lds_b128 %0, %1, off"
               :: "v"(laddr), "v"((unsigned long long)gsrc)
               : "memory");
}

static __device__ inline void wait_async0() {
  asm volatile("s_wait_asynccnt 0x0" ::: "memory");
}

// ---------------------------------------------------------------------------
// fp32 -> bf16 conversion
// ---------------------------------------------------------------------------
__global__ void cvt_f32_bf16(const float* __restrict__ in,
                             bf16_t* __restrict__ out, int n) {
  int i = blockIdx.x * blockDim.x + threadIdx.x;
  if (i < n) out[i] = (bf16_t)in[i];
}

// ---------------------------------------------------------------------------
// GEMM: C[m][c] = sum_k A[m][k] * W[c][k]  (i.e. A @ W^T), bias + scale fused.
// mode 0: Q -> bf16 [b,h,s,64], v=(acc+bias)*scale (scale=1/sqrt(depth))
// mode 1: K -> bf16 [b,h,s,64]
// mode 2: V -> bf16 [b,h,64,s]   (transposed per head)
// mode 3: fp32 [m][c] to d_out
// Block: 256 threads = 8 waves; wave computes 16(M) x 64(N); grid (32, 8).
// ---------------------------------------------------------------------------
__global__ __launch_bounds__(256) void gemm_bf16(
    const bf16_t* __restrict__ A, const bf16_t* __restrict__ W,
    const float* __restrict__ bias, float scale, int mode,
    bf16_t* __restrict__ outb, float* __restrict__ outf) {
  const int lane = threadIdx.x & 31;
  const int wave = threadIdx.x >> 5;
  const int qh   = lane >> 4;      // half-wave
  const int l15  = lane & 15;
  const int mbase = blockIdx.x * 128 + wave * 16;
  const int nbase = blockIdx.y * 64;
  const int arow  = mbase + l15;

  v8f acc[4] = {};
  for (int k0 = 0; k0 < D_MODEL; k0 += 32) {
    ABf a;   // A fragment: lane row = arow, halves select K chunk
    const bf16_t* ap = A + (size_t)arow * D_MODEL + k0 + qh * 16;
    a.h[0] = ld8(ap);
    a.h[1] = ld8(ap + 8);
#pragma unroll
    for (int t = 0; t < 4; ++t) {
      ABf b;  // B fragment: lane col = nbase+t*16+l15, K contiguous per lane
      const bf16_t* wp = W + (size_t)(nbase + t * 16 + l15) * D_MODEL + k0 + qh * 16;
      b.h[0] = ld8(wp);
      b.h[1] = ld8(wp + 8);
      acc[t] = __builtin_amdgcn_wmma_f32_16x16x32_bf16(
          false, a.v, false, b.v, (short)0, acc[t], false, false);
    }
  }

#pragma unroll
  for (int t = 0; t < 4; ++t) {
    const int c  = nbase + t * 16 + l15;
    const float bc = bias[c];
    const int hh = c >> 6, d = c & 63;
#pragma unroll
    for (int r = 0; r < 8; ++r) {
      const int m = mbase + r + 8 * qh;     // C tile: VGPR r, half -> row
      const float v = (acc[t][r] + bc) * scale;
      if (mode == 3) {
        outf[(size_t)m * D_MODEL + c] = v;
      } else {
        const int bb = m >> 11, s = m & (SEQ - 1);
        size_t idx;
        if (mode == 2) idx = ((size_t)(bb * HEADS + hh) * DEPTH + d) * SEQ + s;
        else           idx = ((size_t)(bb * HEADS + hh) * SEQ + s) * DEPTH + d;
        outb[idx] = (bf16_t)v;
      }
    }
  }
}

// ---------------------------------------------------------------------------
// Cooperative async staging of one 32-key block: K tile [32][64] and
// V tile [64][32] (bf16). 512 x 16B chunks over 256 threads: one K chunk and
// one V chunk per thread.
// ---------------------------------------------------------------------------
static __device__ inline void stage_kv(const bf16_t* __restrict__ Kp,
                                       const bf16_t* __restrict__ Vp,
                                       bf16_t* sKbuf, bf16_t* sVbuf,
                                       int kb, int tid) {
  const int r  = tid >> 3, j  = tid & 7;   // K: 32 rows x 8 chunks
  async_cp16(Kp + (size_t)(kb + r) * DEPTH + j * 8, sKbuf + r * DEPTH + j * 8);
  const int d  = tid >> 2, j2 = tid & 3;   // V: 64 rows x 4 chunks
  async_cp16(Vp + (size_t)d * SEQ + kb + j2 * 8,   sVbuf + d * 32 + j2 * 8);
}

// ---------------------------------------------------------------------------
// Flash attention. Wave handles one 16-query tile, streams 32-key blocks
// staged in double-buffered LDS via global_load_async_to_lds_b128.
// Computes S^T = K_block @ Q^T so that the score C-tile layout (lane=query,
// VGPR r + half = key) is exactly the A-fragment layout needed for P @ V.
// ---------------------------------------------------------------------------
__global__ __launch_bounds__(256) void attn_flash(
    const bf16_t* __restrict__ Qh, const bf16_t* __restrict__ Kh,
    const bf16_t* __restrict__ Vt, bf16_t* __restrict__ Oc) {
  __shared__ __align__(16) bf16_t sK[2][32 * DEPTH];   // 8 KB
  __shared__ __align__(16) bf16_t sV[2][DEPTH * 32];   // 8 KB
  const int tid  = threadIdx.x;
  const int lane = tid & 31;
  const int wave = tid >> 5;
  const int qh   = lane >> 4;
  const int l15  = lane & 15;
  const int g    = blockIdx.x * 8 + wave;   // global 16-query tile id
  const int bh   = g >> 7;                  // 128 tiles per (b,h); same for WG
  const int qbase = (g & 127) * 16;

  const bf16_t* Qp = Qh + (size_t)bh * SEQ * DEPTH;
  const bf16_t* Kp = Kh + (size_t)bh * SEQ * DEPTH;
  const bf16_t* Vp = Vt + (size_t)bh * DEPTH * SEQ;

  // Q as B-fragments of S^T: lane col = query, per-lane depth contiguous
  ABf bq[2];
#pragma unroll
  for (int ks = 0; ks < 2; ++ks) {
    const bf16_t* qp = Qp + (size_t)(qbase + l15) * DEPTH + ks * 32 + qh * 16;
    bq[ks].h[0] = ld8(qp);
    bq[ks].h[1] = ld8(qp + 8);
  }

  v8f o[4] = {};
  float mrow = -__builtin_inff();
  float lrow = 0.f;

  // prologue: stage block 0 into buffer 0
  stage_kv(Kp, Vp, sK[0], sV[0], 0, tid);
  wait_async0();
  __syncthreads();

  for (int s = 0; s < SEQ / 32; ++s) {
    const int kb  = s * 32;
    const int cur = s & 1;
    if (s + 1 < SEQ / 32) {
      stage_kv(Kp, Vp, sK[cur ^ 1], sV[cur ^ 1], kb + 32, tid);
    }
    if (kb + 64 < SEQ) {  // warm L2 two blocks ahead (global_prefetch_b8)
      __builtin_prefetch(Kp + (size_t)(kb + 64 + lane) * DEPTH, 0, 1);
    }

    // scores: sc[mt] covers keys kb+mt*16 .. +15 (x 16 queries), from LDS
    v8f sc[2] = {};
#pragma unroll
    for (int mt = 0; mt < 2; ++mt) {
#pragma unroll
      for (int ks = 0; ks < 2; ++ks) {
        ABf a;  // K-block A fragment: lane row = key
        const bf16_t* kp = &sK[cur][(mt * 16 + l15) * DEPTH + ks * 32 + qh * 16];
        a.h[0] = ld8(kp);
        a.h[1] = ld8(kp + 8);
        sc[mt] = __builtin_amdgcn_wmma_f32_16x16x32_bf16(
            false, a.v, false, bq[ks].v, (short)0, sc[mt], false, false);
      }
    }
    // online softmax; per lane: query = l15, keys spread over VGPRs + halves
    float bm = -__builtin_inff();
#pragma unroll
    for (int r = 0; r < 8; ++r) bm = fmaxf(bm, fmaxf(sc[0][r], sc[1][r]));
    bm = fmaxf(bm, __shfl_xor(bm, 16, 32));
    const float mnew = fmaxf(mrow, bm);
    const float corr = __expf(mrow - mnew);
    float p0[8], p1[8];
    float rs = 0.f;
#pragma unroll
    for (int r = 0; r < 8; ++r) {
      p0[r] = __expf(sc[0][r] - mnew);
      p1[r] = __expf(sc[1][r] - mnew);
      rs += p0[r] + p1[r];
    }
    rs += __shfl_xor(rs, 16, 32);
    lrow = lrow * corr + rs;
    mrow = mnew;
    // rescale O accumulators: O rows live at (VGPR r, half) = query r+8*half
#pragma unroll
    for (int r = 0; r < 8; ++r) {
      const float cr = __shfl(corr, r + 8 * qh, 32);
      o[0][r] *= cr; o[1][r] *= cr; o[2][r] *= cr; o[3][r] *= cr;
    }
    // P is already in A-fragment order: elements 0..7 = p0, 8..15 = p1
    v16bf pa;
#pragma unroll
    for (int j = 0; j < 8; ++j) {
      pa[j]     = (bf16_t)p0[j];
      pa[j + 8] = (bf16_t)p1[j];
    }
    // V B-fragments from LDS tile [depth][32 keys]: keys 8h..8h+7, 16+8h..16+8h+7 per lane
#pragma unroll
    for (int dt = 0; dt < 4; ++dt) {
      ABf vb;
      const bf16_t* vp = &sV[cur][(dt * 16 + l15) * 32 + qh * 8];
      vb.h[0] = ld8(vp);
      vb.h[1] = ld8(vp + 16);
      o[dt] = __builtin_amdgcn_wmma_f32_16x16x32_bf16(
          false, pa, false, vb.v, (short)0, o[dt], false, false);
    }

    // publish next buffer / protect current buffer: async writes done, then WG
    wait_async0();
    __syncthreads();
  }

  // normalize and store into [b, s, h*64+d] bf16 for the output projection
  const float inv = 1.0f / lrow;
  const int b = bh >> 3, hh = bh & 7;
#pragma unroll
  for (int r = 0; r < 8; ++r) {
    const float ir = __shfl(inv, r + 8 * qh, 32);
    const int q = qbase + r + 8 * qh;
    bf16_t* op = Oc + (size_t)(b * SEQ + q) * D_MODEL + hh * DEPTH;
#pragma unroll
    for (int dt = 0; dt < 4; ++dt) {
      op[dt * 16 + l15] = (bf16_t)(o[dt][r] * ir);
    }
  }
}

// ---------------------------------------------------------------------------
extern "C" void kernel_launch(void* const* d_in, const int* in_sizes, int n_in,
                              void* d_out, int out_size, void* d_ws, size_t ws_size,
                              hipStream_t stream) {
  (void)in_sizes; (void)n_in; (void)out_size; (void)ws_size;
  const float* x  = (const float*)d_in[0];
  const float* Wq = (const float*)d_in[1];
  const float* bq = (const float*)d_in[2];
  const float* Wk = (const float*)d_in[3];
  const float* bk = (const float*)d_in[4];
  const float* Wv = (const float*)d_in[5];
  const float* bvp= (const float*)d_in[6];
  const float* Wo = (const float*)d_in[7];
  const float* bo = (const float*)d_in[8];
  float* out = (float*)d_out;

  const size_t NX = (size_t)NTOK * D_MODEL;       // 2,097,152
  const size_t NW = (size_t)D_MODEL * D_MODEL;    //   262,144
  bf16_t* xb  = (bf16_t*)d_ws;
  bf16_t* wqb = xb  + NX;
  bf16_t* wkb = wqb + NW;
  bf16_t* wvb = wkb + NW;
  bf16_t* wob = wvb + NW;
  bf16_t* Qhb = wob + NW;
  bf16_t* Khb = Qhb + NX;
  bf16_t* Vtb = Khb + NX;
  bf16_t* Ocb = Vtb + NX;   // total ~22 MB of workspace

  cvt_f32_bf16<<<(int)((NX + 255) / 256), 256, 0, stream>>>(x,  xb,  (int)NX);
  cvt_f32_bf16<<<(int)((NW + 255) / 256), 256, 0, stream>>>(Wq, wqb, (int)NW);
  cvt_f32_bf16<<<(int)((NW + 255) / 256), 256, 0, stream>>>(Wk, wkb, (int)NW);
  cvt_f32_bf16<<<(int)((NW + 255) / 256), 256, 0, stream>>>(Wv, wvb, (int)NW);
  cvt_f32_bf16<<<(int)((NW + 255) / 256), 256, 0, stream>>>(Wo, wob, (int)NW);

  dim3 ggrid(NTOK / 128, D_MODEL / 64);   // (32, 8)
  const float qscale = 0.125f;            // 1/sqrt(64)
  gemm_bf16<<<ggrid, 256, 0, stream>>>(xb, wqb, bq, qscale, 0, Qhb, nullptr);
  gemm_bf16<<<ggrid, 256, 0, stream>>>(xb, wkb, bk, 1.0f,   1, Khb, nullptr);
  gemm_bf16<<<ggrid, 256, 0, stream>>>(xb, wvb, bvp,1.0f,   2, Vtb, nullptr);

  // B*H*S/16 = 2048 query tiles, 8 waves per block -> 256 blocks
  attn_flash<<<256, 256, 0, stream>>>(Qhb, Khb, Vtb, Ocb);

  gemm_bf16<<<ggrid, 256, 0, stream>>>(Ocb, wob, bo, 1.0f, 3, nullptr, out);
}
